// TreeEnergyLossBinary_88802743812178
// MI455X (gfx1250) — matrix operand order repaired and checked by reading, not compiled, not run on hardware
//
#include <hip/hip_runtime.h>
#include <hip/hip_bf16.h>

#define Bb   4
#define Hh   160
#define Wo   160
#define Nn   25600      // Hh*Wo
#define HS   640
#define WS   640
#define CHI  256
#define SIG  0.002f
#define WT   0.4f

typedef __attribute__((ext_vector_type(16))) __bf16 v16bf;
typedef __attribute__((ext_vector_type(8)))  float  v8f;

// ---------------------------------------------------------------- init accum
__global__ void k_init(float* accum) {
    accum[0] = 0.f;   // loss sum
    accum[1] = 0.f;   // roi sum
}

// ------------------------------------------------- bilinear 640->160, 3 chan
__global__ void k_resize_low(const float* __restrict__ lf, float* __restrict__ lowlr) {
    int i = blockIdx.x * blockDim.x + threadIdx.x;
    if (i >= Bb * 3 * Nn) return;
    int bc = i / Nn;                 // b*3 + c
    int t  = i % Nn;
    int y  = t / Wo, x = t % Wo;
    float ys = fminf(fmaxf((y + 0.5f) * ((float)HS / Hh) - 0.5f, 0.f), HS - 1.f);
    float xs = fminf(fmaxf((x + 0.5f) * ((float)WS / Wo) - 0.5f, 0.f), WS - 1.f);
    int y0 = (int)floorf(ys), x0 = (int)floorf(xs);
    int y1 = min(y0 + 1, HS - 1), x1 = min(x0 + 1, WS - 1);
    float wy = ys - y0, wx = xs - x0;
    const float* p = lf + (size_t)bc * HS * WS;
    float v00 = p[y0 * WS + x0], v01 = p[y0 * WS + x1];
    float v10 = p[y1 * WS + x0], v11 = p[y1 * WS + x1];
    float top = v00 * (1.f - wx) + v01 * wx;
    float bot = v10 * (1.f - wx) + v11 * wx;
    lowlr[i] = top * (1.f - wy) + bot * wy;
}

// -------------------------------------------- 3-channel tree edge weights
__global__ void k_wgt_low(const float* __restrict__ lowlr, const int* __restrict__ order,
                          const int* __restrict__ parent, float* __restrict__ wgt) {
    int i = blockIdx.x * blockDim.x + threadIdx.x;
    if (i >= Bb * Nn) return;
    int b = i / Nn, t = i % Nn;
    const int* ord = order + (size_t)b * Nn;
    int a = ord[t];
    int p = ord[parent[(size_t)b * Nn + t]];
    const float* e = lowlr + (size_t)b * 3 * Nn;
    float s = 0.f;
#pragma unroll
    for (int c = 0; c < 3; ++c) {
        float d = e[(size_t)c * Nn + a] - e[(size_t)c * Nn + p];
        s += d * d;
    }
    wgt[i] = (t == 0) ? 0.f : __expf(-s * (1.0f / SIG));
}

// ------------- 256-channel tree edge weights via v_wmma_f32_16x16x32_bf16
// For a tile of 16 nodes: D = e_self - e_parent (16x256).
// diag(D * D^T) = per-node squared distance. A=D, B=D^T: both fragments are
// gathers from the same LDS tile with one contiguous K order (k = 16*(lane/16) + j).
__global__ __launch_bounds__(32) void k_wgt_high(const float* __restrict__ hf,
                                                 const int* __restrict__ order,
                                                 const int* __restrict__ parent,
                                                 float* __restrict__ wgt) {
    __shared__ float dlds[16][CHI];      // 16 KB diff tile
    __shared__ int   nodeA[16], nodeP[16];
    const int tilesPerB = Nn / 16;
    int b    = blockIdx.x / tilesPerB;
    int tile = blockIdx.x % tilesPerB;
    int lane = threadIdx.x;

    const int* ord = order  + (size_t)b * Nn;
    const int* par = parent + (size_t)b * Nn;
    if (lane < 16) {
        int t = tile * 16 + lane;
        nodeA[lane] = ord[t];
        nodeP[lane] = ord[par[t]];
    }
    __syncthreads();

    const float* hfb = hf + (size_t)b * CHI * Nn;
    // 16 lanes gather 16 nodes within one channel plane (L2-resident), next 16
    // lanes take the next plane.
    for (int idx = lane; idx < 16 * CHI; idx += 32) {
        int m = idx & 15;
        int c = idx >> 4;
        size_t off = (size_t)c * Nn;
        dlds[m][c] = hfb[off + nodeA[m]] - hfb[off + nodeP[m]];
    }
    __syncthreads();

    v8f acc = {};
    const int mrow = lane & 15;
    const int aofs = (lane < 16) ? 0 : 16;   // A 16x32 bf16 lane K-striping (contiguous, same as B)
    const int bofs = (lane < 16) ? 0 : 16;   // B 32x16 bf16 lane K-striping
#pragma unroll
    for (int s = 0; s < CHI / 32; ++s) {
        int k0 = s * 32;
        v16bf a = {}, bm = {};
#pragma unroll
        for (int j = 0; j < 16; ++j) {
            int ka = k0 + aofs + j;
            int kb = k0 + bofs + j;
            a[j]  = (__bf16)dlds[mrow][ka];
            bm[j] = (__bf16)dlds[mrow][kb];
        }
        acc = __builtin_amdgcn_wmma_f32_16x16x32_bf16(false, a, false, bm,
                                                      (short)0, acc, false, false);
    }

    // diagonal of the 16x16 f32 accumulator:
    //   m<8  -> lane m      , vgpr m
    //   m>=8 -> lane m+16   , vgpr m-8
    int r = -1, m = 0;
    if (lane < 8)        { r = lane;      m = lane;      }
    else if (lane >= 24) { r = lane - 24; m = lane - 16; }
    if (r >= 0) {
        float sq = 0.f;
#pragma unroll
        for (int j = 0; j < 8; ++j) sq = (r == j) ? acc[j] : sq;
        int t = tile * 16 + m;
        float w = __expf(-sq * (1.0f / SIG));
        if (t == 0) w = 0.f;
        wgt[(size_t)b * Nn + t] = w;
    }
}

// ------------------------------------------------ LDS-resident tree filter
// One workgroup per batch; 2x25600 f32 state lives in dynamic LDS (200 KB of
// the 320 KB/WGP). Lanes 0/1 each own one independent channel chain.
__device__ __forceinline__ void filter_pass(float* x0, float* x1,
                                            const float* __restrict__ src,
                                            const int* __restrict__ ord,
                                            const int* __restrict__ par,
                                            const float* __restrict__ w,
                                            float* __restrict__ dst, int lane) {
    for (int t = lane; t < Nn; t += 32) {
        x0[t] = src[ord[t]];
        x1[t] = 1.0f;
    }
    __syncthreads();
    if (lane < 2) {
        float* xc = lane ? x1 : x0;
        // up-sweep: child into parent, reverse BFS order
#pragma unroll 4
        for (int t = Nn - 1; t >= 1; --t) {
            int p = par[t];
            xc[p] += w[t] * xc[t];
        }
    }
    __syncthreads();
    if (lane < 2) {
        float* xc = lane ? x1 : x0;
        // down-sweep, in place (par[t] < t so parents are final when read)
#pragma unroll 4
        for (int t = 1; t < Nn; ++t) {
            int p = par[t];
            float ww = w[t];
            float u  = xc[t];
            xc[t] = u + ww * (xc[p] - ww * u);
        }
    }
    __syncthreads();
    for (int t = lane; t < Nn; t += 32) dst[ord[t]] = x0[t] / x1[t];
    __syncthreads();
}

__global__ __launch_bounds__(32) void k_tree_scan(
        const float* __restrict__ preds,
        const int* __restrict__ orderL, const int* __restrict__ parentL,
        const float* __restrict__ wgtL,
        const int* __restrict__ orderH, const int* __restrict__ parentH,
        const float* __restrict__ wgtH,
        float* __restrict__ out1, float* __restrict__ AS) {
    extern __shared__ float sm[];            // 2 * Nn floats
    float* x0 = sm;
    float* x1 = sm + Nn;
    int b = blockIdx.x, lane = threadIdx.x;
    size_t o = (size_t)b * Nn;
    filter_pass(x0, x1, preds + o, orderL + o, parentL + o, wgtL + o, out1 + o, lane);
    filter_pass(x0, x1, out1  + o, orderH + o, parentH + o, wgtH + o, AS   + o, lane);
}

// ------------------------------------------------------------- loss reduce
__global__ void k_loss(const float* __restrict__ preds, const float* __restrict__ AS,
                       const float* __restrict__ rois, float* accum) {
    int i = blockIdx.x * blockDim.x + threadIdx.x;
    float l = 0.f, r = 0.f;
    if (i < Bb * Nn) {
        int b = i / Nn, t = i % Nn;
        int y = t / Wo, x = t % Wo;
        r = rois[(size_t)b * HS * WS + (size_t)(y * (HS / Hh)) * WS + x * (WS / Wo)];
        l = r * fabsf(preds[i] - AS[i]);
    }
#pragma unroll
    for (int off = 16; off > 0; off >>= 1) {
        l += __shfl_down(l, off, 32);
        r += __shfl_down(r, off, 32);
    }
    if ((threadIdx.x & 31) == 0) {
        atomicAdd(&accum[0], l);
        atomicAdd(&accum[1], r);
    }
}

__global__ void k_final(const float* accum, float* out) {
    float L = accum[0], N = accum[1];
    out[0] = WT * ((N > 0.f) ? (L / N) : L);
}

// ---------------------------------------------------------------------------
extern "C" void kernel_launch(void* const* d_in, const int* in_sizes, int n_in,
                              void* d_out, int out_size, void* d_ws, size_t ws_size,
                              hipStream_t stream) {
    const float* preds   = (const float*)d_in[0];
    const float* lowf    = (const float*)d_in[1];
    const float* highf   = (const float*)d_in[2];
    const float* rois    = (const float*)d_in[3];
    const int*   orderL  = (const int*)  d_in[4];
    const int*   parentL = (const int*)  d_in[5];
    const int*   orderH  = (const int*)  d_in[6];
    const int*   parentH = (const int*)  d_in[7];
    float* out = (float*)d_out;

    float* ws     = (float*)d_ws;
    float* lowlr  = ws;                        // Bb*3*Nn
    float* wgtL   = lowlr + (size_t)Bb * 3 * Nn;  // Bb*Nn
    float* wgtH   = wgtL  + (size_t)Bb * Nn;      // Bb*Nn
    float* out1   = wgtH  + (size_t)Bb * Nn;      // Bb*Nn
    float* AS     = out1  + (size_t)Bb * Nn;      // Bb*Nn
    float* accum  = AS    + (size_t)Bb * Nn;      // 2

    k_init<<<1, 1, 0, stream>>>(accum);

    {
        int n = Bb * 3 * Nn;
        k_resize_low<<<(n + 255) / 256, 256, 0, stream>>>(lowf, lowlr);
    }
    {
        int n = Bb * Nn;
        k_wgt_low<<<(n + 255) / 256, 256, 0, stream>>>(lowlr, orderL, parentL, wgtL);
    }
    {
        int blocks = Bb * (Nn / 16);
        k_wgt_high<<<blocks, 32, 0, stream>>>(highf, orderH, parentH, wgtH);
    }
    {
        size_t shmem = (size_t)2 * Nn * sizeof(float);   // 204800 B of 320 KB/WGP
        k_tree_scan<<<Bb, 32, shmem, stream>>>(preds, orderL, parentL, wgtL,
                                               orderH, parentH, wgtH, out1, AS);
    }
    {
        int n = Bb * Nn;
        k_loss<<<(n + 255) / 256, 256, 0, stream>>>(preds, AS, rois, accum);
    }
    k_final<<<1, 1, 0, stream>>>(accum, out);
}
